// MultiHeadSelfAttention_6511170421547
// MI455X (gfx1250) — matrix operand order repaired and checked by reading, not compiled, not run on hardware
//
#include <hip/hip_runtime.h>
#include <hip/hip_bf16.h>

#define EMBED 512
#define SEQ   1024
#define NB    8
#define NH    8
#define HD    64

typedef __attribute__((ext_vector_type(16))) __bf16 v16bf;
typedef __attribute__((ext_vector_type(8)))  float  v8f;
typedef __attribute__((ext_vector_type(4)))  unsigned int u32x4;
typedef __attribute__((ext_vector_type(8)))  int i32x8;
typedef __attribute__((ext_vector_type(4)))  int i32x4;

// ---------------------------------------------------------------------------
// WMMA fragment k order used here (contiguous for A and B; wave32, 16x16x32 bf16):
//   A (16x32): lane l: m = l&15, ka = 16*(l>>4);
//       frag[j] = A[m][ka+j]                                  (j=0..15)
//   B (32x16): lane l: n = l&15, kb = 16*(l>>4);
//       frag[j] = B[kb+j][n]                                  (j=0..15)
//   C/D (16x16 f32): lane l: n = l&15; element i = row (i + 8*(l>>4))
// ---------------------------------------------------------------------------
__device__ __forceinline__ v8f wmma_bf16(const v16bf& a, const v16bf& b, const v8f& c) {
    return __builtin_amdgcn_wmma_f32_16x16x32_bf16(
        false, a, false, b, (short)0, c, false, false);
}

// ---------------------------------------------------------------------------
// TDM: DMA a 2D bf16 tile (tile1 rows x tile0 elems, packed rows) from a
// [dim1 x dim0] row-major tensor (row stride = stride0 elems) into LDS.
// D# bit layout per CDNA5 ISA §8.3/8.4 (group0: count/lds/global/type;
// group1: data_size=2B, tensor dims, tile dims, dim0 stride).
// ---------------------------------------------------------------------------
#if defined(__has_builtin)
#if __has_builtin(__builtin_amdgcn_tensor_load_to_lds)
#define HAVE_TDM 1
#endif
#endif

#ifdef HAVE_TDM
__device__ __forceinline__ void tdm_load_2d_bf16(
    unsigned int lds_addr, const void* gaddr,
    unsigned int dim0, unsigned int dim1, unsigned long long stride0,
    unsigned int tile0, unsigned int tile1)
{
    const unsigned long long ga = (unsigned long long)gaddr;
    u32x4 g0;
    g0[0] = 1u;                                              // count=1 (user mode)
    g0[1] = lds_addr;                                        // LDS byte address
    g0[2] = (unsigned int)ga;                                // global_addr[31:0]
    g0[3] = (unsigned int)((ga >> 32) & 0x1FFFFFFu) | (2u << 30);  // [56:32] | type=2

    i32x8 g1;
    g1[0] = (int)(1u << 16);                                 // data_size=1 -> 2 bytes
    g1[1] = (int)(dim0 << 16);                               // tensor_dim0[15:0] @bits[63:48]
    g1[2] = (int)((dim0 >> 16) | (dim1 << 16));              // dim0[31:16] | dim1[15:0]
    g1[3] = (int)((dim1 >> 16) | (tile0 << 16));             // dim1[31:16] | tile_dim0
    g1[4] = (int)(tile1 & 0xFFFF);                           // tile_dim1 (tile_dim2=0)
    g1[5] = (int)(unsigned int)stride0;                      // dim0_stride[31:0]
    g1[6] = (int)((stride0 >> 32) & 0xFFFF);                 // dim0_stride[47:32]
    g1[7] = 0;                                               // dim1_stride hi (unused, 2D)

    const i32x4 z4 = {0, 0, 0, 0};
#if __clang_major__ >= 23
    const i32x8 z8 = {0, 0, 0, 0, 0, 0, 0, 0};
    __builtin_amdgcn_tensor_load_to_lds(g0, g1, z4, z4, z8, 0);
#else
    __builtin_amdgcn_tensor_load_to_lds(g0, g1, z4, z4, 0);
#endif
}
#endif

// ---------------------------------------------------------------------------
// Kernel 1: fused Q/K/V projection.  xs = raw reshape of x -> [B, S, E].
// Block = 8 waves sharing one LDS-staged 16x512 bf16 x-tile; wave w computes
// e-tile (blockIdx.x*8 + w) of the 96 tiles spanning Wq|Wk|Wv columns.
// Q,K stored bf16 [b*NH+h][s][d] (Q pre-scaled 1/sqrt(d)); V transposed
// [b*NH+h][d][s] so the PV-stage B-fragments are contiguous.
// ---------------------------------------------------------------------------
__global__ __launch_bounds__(256) void mhsa_qkv_proj(
    const float* __restrict__ x,
    const float* __restrict__ Wq, const float* __restrict__ bq,
    const float* __restrict__ Wk, const float* __restrict__ bk,
    const float* __restrict__ Wv, const float* __restrict__ bv,
    __bf16* __restrict__ qb, __bf16* __restrict__ kb, __bf16* __restrict__ vt)
{
    __shared__ __bf16 xt[16 * EMBED];          // 16 KB staged A tile

    const int tid   = threadIdx.x;
    const int b     = blockIdx.z;
    const int sbase = blockIdx.y * 16;

    // ---- cooperative stage: x tile 16x512 fp32 -> bf16 in LDS ----
    {
        const int row = tid >> 4;              // 0..15
        const int col = (tid & 15) * 32;       // 0..480
        const float* src = x + ((size_t)b * SEQ + sbase + row) * EMBED + col;
        __bf16* dst = &xt[row * EMBED + col];
#pragma unroll
        for (int j = 0; j < 32; ++j) dst[j] = (__bf16)src[j];
    }
    __syncthreads();

    const int wave = tid >> 5;
    const int lane = tid & 31;
    const int half = lane >> 4;
    const int lm   = lane & 15;

    const int et    = blockIdx.x * 8 + wave;   // 0..95 over {Wq,Wk,Wv} columns
    const int mat   = et >> 5;                 // 0=Q 1=K 2=V
    const int ebase = (et & 31) * 16;

    const float* W    = (mat == 0) ? Wq : (mat == 1) ? Wk : Wv;
    const float* bias = (mat == 0) ? bq : (mat == 1) ? bk : bv;
    const float scale = (mat == 0) ? 0.125f : 1.0f;

    const __bf16* arow = &xt[lm * EMBED];                     // A row (m = lm)
    const float*  wrow = W + (size_t)(ebase + lm) * EMBED;    // B col (n = lm)

    v8f c = {};
    for (int kk = 0; kk < EMBED; kk += 32) {
        v16bf a, bf;
        const __bf16* ap = arow + kk + half * 16;
        const float*  bp = wrow + kk + half * 16;
#pragma unroll
        for (int j = 0; j < 8; ++j) { a[j] = ap[j]; a[8 + j] = ap[8 + j]; }
#pragma unroll
        for (int j = 0; j < 16; ++j) bf[j] = (__bf16)bp[j];
        c = wmma_bf16(a, bf, c);
    }

    const int e  = ebase + lm;
    const int h  = e >> 6;
    const int dl = e & 63;
    const float bv_ = bias[e];
    const size_t bh = (size_t)b * NH + h;

#pragma unroll
    for (int i = 0; i < 8; ++i) {
        const int m = i + 8 * half;
        const float v = (c[i] + bv_) * scale;
        if (mat < 2) {
            __bf16* dst = (mat == 0) ? qb : kb;
            dst[(bh * SEQ + sbase + m) * HD + dl] = (__bf16)v;
        } else {
            vt[(bh * HD + dl) * SEQ + sbase + m] = (__bf16)v;
        }
    }
}

// ---------------------------------------------------------------------------
// Kernel 2: flash attention. Block = 8 waves = 8 query tiles (128 queries)
// of one (b,h); K/V 32-key chunks DMA'd into LDS by the Tensor Data Mover
// (wave 0 issues TENSOR_LOAD_TO_LDS, waits TENSORcnt, block barrier) and
// shared by all waves. Per chunk/wave: 4 WMMA scores, online softmax
// (__shfl_xor row reductions), wave-private LDS transpose of P, 4 WMMA P*V.
// ---------------------------------------------------------------------------
__global__ __launch_bounds__(256) void mhsa_attn(
    const __bf16* __restrict__ qb, const __bf16* __restrict__ kb,
    const __bf16* __restrict__ vt, float* __restrict__ ctx)
{
    __shared__ __bf16 kt[32 * HD];             // 4 KB  K chunk  [key][d]
    __shared__ __bf16 vts[HD * 32];            // 4 KB  V chunk  [d][key]
    __shared__ __bf16 pt[8][16 * 32];          // 8 KB  per-wave P bounce

    const int tid  = threadIdx.x;
    const int wave = tid >> 5;
    const int lane = tid & 31;
    const int half = lane >> 4;
    const int lm   = lane & 15;

    const int b = blockIdx.z;
    const int h = blockIdx.y;
    const size_t bh = (size_t)b * NH + h;
    const int qbase = (blockIdx.x * 8 + wave) * 16;

    // Q A-fragments for the two d-chunks
    v16bf aq[2];
#pragma unroll
    for (int t = 0; t < 2; ++t) {
        const __bf16* qp = qb + (bh * SEQ + qbase + lm) * HD + t * 32 + half * 16;
#pragma unroll
        for (int j = 0; j < 8; ++j) { aq[t][j] = qp[j]; aq[t][8 + j] = qp[8 + j]; }
    }

    float mrow[8], lrow[8];
#pragma unroll
    for (int i = 0; i < 8; ++i) { mrow[i] = -1e30f; lrow[i] = 0.0f; }
    v8f acc[4] = {};

    const __bf16* kchunk = kb + bh * SEQ * HD;     // [key][d]
    const __bf16* vchunk = vt + bh * HD * SEQ;     // [d][s]

    for (int kbase = 0; kbase < SEQ; kbase += 32) {
        __syncthreads();   // previous chunk's readers done

#ifdef HAVE_TDM
        // ---- TDM DMA: K (32x64) and V^T (64x32) chunks straight into LDS ----
        if (wave == 0) {
            tdm_load_2d_bf16((unsigned int)(unsigned long long)&kt[0],
                             kchunk + (size_t)kbase * HD,
                             /*dim0=*/HD, /*dim1=*/SEQ - kbase, /*stride0=*/HD,
                             /*tile0=*/HD, /*tile1=*/32);
            tdm_load_2d_bf16((unsigned int)(unsigned long long)&vts[0],
                             vchunk + kbase,
                             /*dim0=*/SEQ, /*dim1=*/HD, /*stride0=*/SEQ,
                             /*tile0=*/32, /*tile1=*/HD);
            __builtin_amdgcn_s_wait_tensorcnt(0);
        }
#else
        // ---- fallback: cooperative VGPR staging ----
        {
            const int idx = tid * 8;                       // 2048 bf16 each
            const int key = idx >> 6, d = idx & 63;
            *(uint4*)&kt[idx] = *(const uint4*)(kchunk + (size_t)(kbase + key) * HD + d);
            const int dv = idx >> 5, kv = idx & 31;
            *(uint4*)&vts[idx] = *(const uint4*)(vchunk + (size_t)dv * SEQ + kbase + kv);
        }
#endif
        // prefetch next chunk toward L2 while this one is consumed
        if (kbase + 32 < SEQ) {
            __builtin_prefetch((const void*)(kchunk + (size_t)(kbase + 32 + (tid >> 3)) * HD), 0, 1);
            __builtin_prefetch((const void*)(vchunk + (size_t)(tid >> 2) * SEQ + kbase + 32), 0, 1);
        }
        __syncthreads();   // staged data visible to all waves

        // ---- scores: two 16-key tiles, K=64 over two d-chunks ----
        v8f s0 = {}, s1 = {};
#pragma unroll
        for (int t = 0; t < 2; ++t) {
            v16bf bk0, bk1;
            const __bf16* kp0 = &kt[lm * HD + t * 32 + half * 16];
            const __bf16* kp1 = kp0 + 16 * HD;
#pragma unroll
            for (int j = 0; j < 16; ++j) { bk0[j] = kp0[j]; bk1[j] = kp1[j]; }
            s0 = wmma_bf16(aq[t], bk0, s0);
            s1 = wmma_bf16(aq[t], bk1, s1);
        }

        // ---- online softmax across the 32 keys ----
        float alpha[8];
#pragma unroll
        for (int i = 0; i < 8; ++i) {
            float rm = fmaxf(s0[i], s1[i]);
#pragma unroll
            for (int msk = 1; msk <= 8; msk <<= 1) rm = fmaxf(rm, __shfl_xor(rm, msk, 32));
            const float mnew = fmaxf(mrow[i], rm);
            const float a_   = __expf(mrow[i] - mnew);
            const float p0   = __expf(s0[i] - mnew);
            const float p1   = __expf(s1[i] - mnew);
            s0[i] = p0; s1[i] = p1;
            float rs = p0 + p1;
#pragma unroll
            for (int msk = 1; msk <= 8; msk <<= 1) rs += __shfl_xor(rs, msk, 32);
            lrow[i]  = lrow[i] * a_ + rs;
            mrow[i]  = mnew;
            alpha[i] = a_;
        }
#pragma unroll
        for (int dt = 0; dt < 4; ++dt)
#pragma unroll
            for (int i = 0; i < 8; ++i) acc[dt][i] *= alpha[i];

        // ---- transpose P (C layout -> A layout) via wave-private LDS ----
        __bf16* pw = pt[wave];
#pragma unroll
        for (int i = 0; i < 8; ++i) {
            const int m = i + 8 * half;
            pw[m * 32 + lm]      = (__bf16)s0[i];
            pw[m * 32 + 16 + lm] = (__bf16)s1[i];
        }
        __builtin_amdgcn_wave_barrier();   // wave-private: in-order DS suffices
        v16bf ap;
        const __bf16* pp = &pw[lm * 32 + half * 16];
#pragma unroll
        for (int j = 0; j < 8; ++j) { ap[j] = pp[j]; ap[8 + j] = pp[8 + j]; }

        // ---- acc += P @ V ----
#pragma unroll
        for (int dt = 0; dt < 4; ++dt) {
            v16bf bvf;
            const __bf16* vp = &vts[(dt * 16 + lm) * 32 + half * 16];
#pragma unroll
            for (int j = 0; j < 16; ++j) bvf[j] = vp[j];
            acc[dt] = wmma_bf16(ap, bvf, acc[dt]);
        }
    }

    // ---- normalize & store ctx[b][s][e] (fp32) ----
#pragma unroll
    for (int i = 0; i < 8; ++i) {
        const float inv = 1.0f / lrow[i];
        const int m = i + 8 * half;
#pragma unroll
        for (int dt = 0; dt < 4; ++dt)
            ctx[((size_t)b * SEQ + qbase + m) * EMBED + h * HD + dt * 16 + lm] = acc[dt][i] * inv;
    }
}

// ---------------------------------------------------------------------------
// Kernel 3: output projection with fused [B,S,C]->[B,C,S] permute.
// Block = 8 waves x 4 e-tiles covering all 512 output channels of one s-tile;
// ctx tile 16x512 staged once in LDS (read exactly once from memory).
// M = e-tile / N = s-tile so final stores are lane-coalesced along s.
// ---------------------------------------------------------------------------
__global__ __launch_bounds__(256) void mhsa_out_proj(
    const float* __restrict__ ctx, const float* __restrict__ Wo,
    const float* __restrict__ bo, float* __restrict__ out)
{
    __shared__ __bf16 ct[16 * EMBED];          // 16 KB staged ctx tile (B matrix)

    const int tid   = threadIdx.x;
    const int sbase = blockIdx.x * 16;
    const int b     = blockIdx.y;

    {
        const int row = tid >> 4;
        const int col = (tid & 15) * 32;
        const float* src = ctx + ((size_t)b * SEQ + sbase + row) * EMBED + col;
        __bf16* dst = &ct[row * EMBED + col];
#pragma unroll
        for (int j = 0; j < 32; ++j) dst[j] = (__bf16)src[j];
    }
    __syncthreads();

    const int wave = tid >> 5;
    const int lane = tid & 31;
    const int half = lane >> 4;
    const int lm   = lane & 15;

    for (int sub = 0; sub < 4; ++sub) {
        const int ebase = (wave * 4 + sub) * 16;
        const float* wrow = Wo + (size_t)(ebase + lm) * EMBED;    // A row (m = lm)
        const __bf16* crow = &ct[lm * EMBED];                     // B col (n = lm)

        v8f c = {};
        for (int kk = 0; kk < EMBED; kk += 32) {
            v16bf aw, bc;
            const float*  ap = wrow + kk + half * 16;
            const __bf16* bp = crow + kk + half * 16;
#pragma unroll
            for (int j = 0; j < 8; ++j) {
                aw[j]     = (__bf16)ap[j];
                aw[8 + j] = (__bf16)ap[8 + j];
            }
#pragma unroll
            for (int j = 0; j < 16; ++j) bc[j] = bp[j];
            c = wmma_bf16(aw, bc, c);
        }

#pragma unroll
        for (int i = 0; i < 8; ++i) {
            const int e = ebase + i + 8 * half;
            out[((size_t)b * EMBED + e) * SEQ + sbase + lm] = c[i] + bo[e];
        }
    }
}

// ---------------------------------------------------------------------------
extern "C" void kernel_launch(void* const* d_in, const int* in_sizes, int n_in,
                              void* d_out, int out_size, void* d_ws, size_t ws_size,
                              hipStream_t stream) {
    const float* x  = (const float*)d_in[0];
    const float* Wq = (const float*)d_in[1];
    const float* bq = (const float*)d_in[2];
    const float* Wk = (const float*)d_in[3];
    const float* bk = (const float*)d_in[4];
    const float* Wv = (const float*)d_in[5];
    const float* bv = (const float*)d_in[6];
    const float* Wo = (const float*)d_in[7];
    const float* bo = (const float*)d_in[8];

    char* ws = (char*)d_ws;
    __bf16* qb = (__bf16*)(ws);                         //  8 MB: Q  [b,h,s,d] bf16
    __bf16* kb = (__bf16*)(ws + ((size_t)8  << 20));    //  8 MB: K  [b,h,s,d] bf16
    __bf16* vt = (__bf16*)(ws + ((size_t)16 << 20));    //  8 MB: V^T[b,h,d,s] bf16
    float*  cx = (float* )(ws + ((size_t)24 << 20));    // 16 MB: ctx[b,s,e]  f32

    dim3 g1(96 / 8, SEQ / 16, NB);
    mhsa_qkv_proj<<<g1, 256, 0, stream>>>(x, Wq, bq, Wk, bk, Wv, bv, qb, kb, vt);

    dim3 g2(SEQ / 128, NH, NB);
    mhsa_attn<<<g2, 256, 0, stream>>>(qb, kb, vt, cx);

    dim3 g3(SEQ / 16, NB);
    mhsa_out_proj<<<g3, 256, 0, stream>>>(cx, Wo, bo, (float*)d_out);
}
